// HeteroGNNLayer_47493748359690
// MI455X (gfx1250) — matrix operand order repaired and checked by reading, not compiled, not run on hardware
//
#include <hip/hip_runtime.h>
#include <stdint.h>

#define NA   50000
#define NB   50000
#define ETOT 500000
#define DD   128
#define FSTR 392   // feat row stride in ushorts: 384 data + 8 pad (16B-aligned rows, bank-skewed)
#define EW   4     // waves per block

typedef __attribute__((ext_vector_type(8)))  unsigned short v8u;
typedef __attribute__((ext_vector_type(16))) unsigned short v16u;
typedef __attribute__((ext_vector_type(16))) __bf16         v16bf;
typedef __attribute__((ext_vector_type(8)))  float          v8f;

union FragAB { v16u u; v16bf b; };

__device__ __forceinline__ unsigned short f2bf(float f) {
  unsigned int u = __float_as_uint(f);
  u += 0x7FFFu + ((u >> 16) & 1u);          // round-to-nearest-even
  return (unsigned short)(u >> 16);
}
__device__ __forceinline__ float bf2f(unsigned short h) {
  return __uint_as_float(((unsigned int)h) << 16);
}
__device__ __forceinline__ v16u join8(v8u lo, v8u hi) {
  return __builtin_shufflevector(lo, hi, 0,1,2,3,4,5,6,7,8,9,10,11,12,13,14,15);
}
__device__ __forceinline__ v8f wmma_bf16(v16u a, v16u b, v8f c) {
  FragAB fa, fb; fa.u = a; fb.u = b;
  return __builtin_amdgcn_wmma_f32_16x16x32_bf16(false, fa.b, false, fb.b,
                                                 (short)0, c, false, false);
}
// A fragment (16x32 bf16): lane holds row (lane&15); elems 0..7 -> K=kb*32+lo+i,
// elems 8..15 -> K=kb*32+16+lo+i, lo = 8*(lane>=16).  rowp = that lane's row.
__device__ __forceinline__ v16u loadA(const unsigned short* rowp, int kb, int hf) {
  const v8u* p0 = (const v8u*)(rowp + kb*32 + hf*8);
  const v8u* p1 = (const v8u*)(rowp + kb*32 + 16 + hf*8);
  return join8(*p0, *p1);
}
// B fragment (32x16) for out = X @ W^T with row-major bf16 W[N][K]: B[k][n] = W[n][k].
// elems 0..7 -> K = kb*32+lo+e, elems 8..15 -> K = kb*32+16+lo+(e-8), lo = 8*(lane>=16)
__device__ __forceinline__ v16u loadB(const unsigned short* W, int K, int nb, int kb, int lane) {
  const int n = nb*16 + (lane & 15);
  const int k = kb*32 + (lane >> 4)*8;
  const v8u* p = (const v8u*)(W + (size_t)n*K + k);
  return join8(p[0], p[2]);
}

#define SWZ_ADD(x, m) \
  (x += __int_as_float(__builtin_amdgcn_ds_swizzle(__float_as_int(x), ((m) << 10) | 0x1f)))

__global__ void cvt_kernel(const float* __restrict__ in, unsigned short* __restrict__ out, int n) {
  int i = blockIdx.x * 256 + threadIdx.x;
  if (i < n) out[i] = f2bf(in[i]);
}

// Fused per-edge pipeline: gather -> ep/ec (WMMA) -> feat(LDS) -> MLP (WMMA)
// -> sigmoid edge weight -> atomic scatter of w*par into msg.
__global__ __launch_bounds__(128) void edge_kernel(
    const unsigned short* __restrict__ xsrc, const unsigned short* __restrict__ xdst,
    const int* __restrict__ ei,
    const unsigned short* __restrict__ Wp, const float* __restrict__ bp,
    const unsigned short* __restrict__ Wc, const float* __restrict__ bc,
    const unsigned short* __restrict__ W1, const float* __restrict__ b1,
    const float* __restrict__ W2, const float* __restrict__ b2,
    float* __restrict__ msg, int nTiles)
{
  __shared__ unsigned short feat[EW][16 * FSTR];
  __shared__ float wsig[EW][16];
  const int lane = threadIdx.x & 31;
  const int wv   = threadIdx.x >> 5;
  const int tile = blockIdx.x * EW + wv;
  if (tile >= nTiles) return;                 // wave-uniform exit; no cross-wave deps
  const int rr = lane & 15;                   // edge row (A) / N column (C)
  const int hf = lane >> 4;
  const int e  = tile * 16 + rr;
  const int si = ei[e];
  const int di = ei[ETOT + e];
  const unsigned short* prow = xsrc + (size_t)si * DD;
  const unsigned short* crow = xdst + (size_t)di * DD;
  unsigned short* fw = &feat[wv][0];

  v16u aP[4], aC[4];
#pragma unroll
  for (int kb = 0; kb < 4; ++kb) { aP[kb] = loadA(prow, kb, hf); aC[kb] = loadA(crow, kb, hf); }

  // |par - cld| -> feat[:, 256:384]
#pragma unroll
  for (int kb = 0; kb < 4; ++kb) {
#pragma unroll
    for (int i = 0; i < 16; ++i) {
      const int K = kb*32 + ((i < 8) ? (hf*8 + i) : (16 + hf*8 + (i - 8)));
      fw[rr*FSTR + 256 + K] = f2bf(fabsf(bf2f(aP[kb][i]) - bf2f(aC[kb][i])));
    }
  }

  // ep = lrelu(par@Wp^T+bp) -> feat[:,0:128]; ec -> feat[:,128:256]
  for (int nb = 0; nb < 8; ++nb) {
    v8f cp = {}, cc = {};
#pragma unroll
    for (int kb = 0; kb < 4; ++kb) {
      cp = wmma_bf16(aP[kb], loadB(Wp, DD, nb, kb, lane), cp);
      cc = wmma_bf16(aC[kb], loadB(Wc, DD, nb, kb, lane), cc);
    }
    const int col = nb*16 + rr;
    const float bP = bp[col], bC = bc[col];
#pragma unroll
    for (int m = 0; m < 8; ++m) {
      const int r = m + hf*8;                 // C frag: M = m + 8*hf, N = rr
      float v = cp[m] + bP; v = (v > 0.f) ? v : 0.01f * v;
      fw[r*FSTR + col] = f2bf(v);
      float u = cc[m] + bC; u = (u > 0.f) ? u : 0.01f * u;
      fw[r*FSTR + 128 + col] = f2bf(u);
    }
  }

  __builtin_amdgcn_fence(__ATOMIC_ACQ_REL, "workgroup");
  __builtin_amdgcn_wave_barrier();

  // h = relu(feat @ W1^T + b1); fold W2 dot into the epilogue
  float sacc[8] = {0.f,0.f,0.f,0.f,0.f,0.f,0.f,0.f};
  for (int nb = 0; nb < 8; ++nb) {
    v8f ch = {};
#pragma unroll
    for (int kb = 0; kb < 12; ++kb) {
      const v8u* q0 = (const v8u*)(fw + rr*FSTR + kb*32 + hf*8);
      const v8u* q1 = (const v8u*)(fw + rr*FSTR + kb*32 + 16 + hf*8);
      ch = wmma_bf16(join8(*q0, *q1), loadB(W1, 3*DD, nb, kb, lane), ch);
    }
    const int col = nb*16 + rr;
    const float bb = b1[col], w2 = W2[col];
#pragma unroll
    for (int m = 0; m < 8; ++m) {
      float h = ch[m] + bb; h = (h > 0.f) ? h : 0.f;
      sacc[m] += h * w2;
    }
  }
  // butterfly sum over the 16 lanes of each half (rows 0-7 low, 8-15 high)
#pragma unroll
  for (int m = 0; m < 8; ++m) {
    SWZ_ADD(sacc[m], 1); SWZ_ADD(sacc[m], 2); SWZ_ADD(sacc[m], 4); SWZ_ADD(sacc[m], 8);
  }
  if (rr == 0) {
    const float b2v = b2[0];
#pragma unroll
    for (int m = 0; m < 8; ++m)
      wsig[wv][hf*8 + m] = 1.f / (1.f + __expf(-(sacc[m] + b2v)));
  }
  __builtin_amdgcn_fence(__ATOMIC_ACQ_REL, "workgroup");
  __builtin_amdgcn_wave_barrier();
  const float w = wsig[wv][rr];

  // scatter w * par into msg[dst]
  float* mrow = msg + (size_t)di * DD;
#pragma unroll
  for (int kb = 0; kb < 4; ++kb) {
#pragma unroll
    for (int i = 0; i < 16; ++i) {
      const int K = kb*32 + ((i < 8) ? (hf*8 + i) : (16 + hf*8 + (i - 8)));
      (void)__hip_atomic_fetch_add(&mrow[K], bf2f(aP[kb][i]) * w,
                                   __ATOMIC_RELAXED, __HIP_MEMORY_SCOPE_AGENT);
    }
  }
}

// out = msg @ Wrel^T + x @ Wroot^T + brel + broot
__global__ __launch_bounds__(128) void node_kernel(
    const float* __restrict__ msg, const unsigned short* __restrict__ xn,
    const unsigned short* __restrict__ Wrel,  const float* __restrict__ brel,
    const unsigned short* __restrict__ Wroot, const float* __restrict__ broot,
    float* __restrict__ out, int nTiles)
{
  const int lane = threadIdx.x & 31;
  const int wv   = threadIdx.x >> 5;
  const int tile = blockIdx.x * EW + wv;
  if (tile >= nTiles) return;
  const int rr = lane & 15, hf = lane >> 4;
  const size_t node = (size_t)tile * 16 + rr;
  const float* mrow = msg + node * DD;
  const unsigned short* xrow = xn + node * DD;

  v16u aM[4], aX[4];
#pragma unroll
  for (int kb = 0; kb < 4; ++kb) {
    aX[kb] = loadA(xrow, kb, hf);
    const float4* p0 = (const float4*)(mrow + kb*32 + hf*8);
    const float4* p1 = (const float4*)(mrow + kb*32 + 16 + hf*8);
    float4 q0 = p0[0], q1 = p0[1], q2 = p1[0], q3 = p1[1];
    v16u t;
    t[0]=f2bf(q0.x);  t[1]=f2bf(q0.y);  t[2]=f2bf(q0.z);  t[3]=f2bf(q0.w);
    t[4]=f2bf(q1.x);  t[5]=f2bf(q1.y);  t[6]=f2bf(q1.z);  t[7]=f2bf(q1.w);
    t[8]=f2bf(q2.x);  t[9]=f2bf(q2.y);  t[10]=f2bf(q2.z); t[11]=f2bf(q2.w);
    t[12]=f2bf(q3.x); t[13]=f2bf(q3.y); t[14]=f2bf(q3.z); t[15]=f2bf(q3.w);
    aM[kb] = t;
  }
  for (int nb = 0; nb < 8; ++nb) {
    v8f c = {};
#pragma unroll
    for (int kb = 0; kb < 4; ++kb) c = wmma_bf16(aM[kb], loadB(Wrel,  DD, nb, kb, lane), c);
#pragma unroll
    for (int kb = 0; kb < 4; ++kb) c = wmma_bf16(aX[kb], loadB(Wroot, DD, nb, kb, lane), c);
    const int col = nb*16 + rr;
    const float bias = brel[col] + broot[col];
#pragma unroll
    for (int m = 0; m < 8; ++m)
      out[((size_t)tile*16 + m + hf*8) * DD + col] = c[m] + bias;
  }
}

extern "C" void kernel_launch(void* const* d_in, const int* in_sizes, int n_in,
                              void* d_out, int out_size, void* d_ws, size_t ws_size,
                              hipStream_t stream) {
  (void)in_sizes; (void)n_in; (void)out_size; (void)ws_size;
  const float* x_a = (const float*)d_in[0];
  const float* x_b = (const float*)d_in[1];
  const float* Wp  = (const float*)d_in[2];
  const float* bp  = (const float*)d_in[3];
  const float* Wc  = (const float*)d_in[4];
  const float* bc  = (const float*)d_in[5];
  const float* W1  = (const float*)d_in[6];
  const float* b1  = (const float*)d_in[7];
  const float* W2  = (const float*)d_in[8];
  const float* b2  = (const float*)d_in[9];
  const float* Wrel_ab  = (const float*)d_in[10];
  const float* brel_ab  = (const float*)d_in[11];
  const float* Wroot_ab = (const float*)d_in[12];
  const float* broot_ab = (const float*)d_in[13];
  const float* Wrel_ba  = (const float*)d_in[14];
  const float* brel_ba  = (const float*)d_in[15];
  const float* Wroot_ba = (const float*)d_in[16];
  const float* broot_ba = (const float*)d_in[17];
  const int* ei_ab = (const int*)d_in[18];
  const int* ei_ba = (const int*)d_in[19];

  // workspace layout (all section sizes multiples of 256B)
  char* p = (char*)d_ws;
  unsigned short* xaB  = (unsigned short*)p; p += (size_t)NA*DD*2;
  unsigned short* xbB  = (unsigned short*)p; p += (size_t)NB*DD*2;
  unsigned short* WpB  = (unsigned short*)p; p += DD*DD*2;
  unsigned short* WcB  = (unsigned short*)p; p += DD*DD*2;
  unsigned short* W1B  = (unsigned short*)p; p += DD*3*DD*2;
  unsigned short* WrelABB  = (unsigned short*)p; p += DD*DD*2;
  unsigned short* WrootABB = (unsigned short*)p; p += DD*DD*2;
  unsigned short* WrelBAB  = (unsigned short*)p; p += DD*DD*2;
  unsigned short* WrootBAB = (unsigned short*)p; p += DD*DD*2;
  float* msgB = (float*)p; p += (size_t)NB*DD*4;
  float* msgA = (float*)p; p += (size_t)NA*DD*4;

  hipMemsetAsync(msgB, 0, (size_t)NB*DD*4, stream);
  hipMemsetAsync(msgA, 0, (size_t)NA*DD*4, stream);

  auto cvt = [&](const float* in, unsigned short* out, int n) {
    cvt_kernel<<<(n + 255) / 256, 256, 0, stream>>>(in, out, n);
  };
  cvt(x_a, xaB, NA*DD);
  cvt(x_b, xbB, NB*DD);
  cvt(Wp, WpB, DD*DD);
  cvt(Wc, WcB, DD*DD);
  cvt(W1, W1B, DD*3*DD);
  cvt(Wrel_ab,  WrelABB,  DD*DD);
  cvt(Wroot_ab, WrootABB, DD*DD);
  cvt(Wrel_ba,  WrelBAB,  DD*DD);
  cvt(Wroot_ba, WrootBAB, DD*DD);

  const int eTiles  = ETOT / 16;                       // 31250
  const int eBlocks = (eTiles + EW - 1) / EW;
  // A->B edges: src = x_a, dst = x_b, accumulate into msgB
  edge_kernel<<<eBlocks, 32*EW, 0, stream>>>(xaB, xbB, ei_ab, WpB, bp, WcB, bc,
                                             W1B, b1, W2, b2, msgB, eTiles);
  // B->A edges: src = x_b, dst = x_a, accumulate into msgA
  edge_kernel<<<eBlocks, 32*EW, 0, stream>>>(xbB, xaB, ei_ba, WpB, bp, WcB, bc,
                                             W1B, b1, W2, b2, msgA, eTiles);

  const int nTiles  = NA / 16;                          // 3125
  const int nBlocks = (nTiles + EW - 1) / EW;
  float* out_a = (float*)d_out;
  float* out_b = (float*)d_out + (size_t)NA * DD;
  node_kernel<<<nBlocks, 32*EW, 0, stream>>>(msgA, xaB, WrelBAB, brel_ba,
                                             WrootBAB, broot_ba, out_a, nTiles);
  node_kernel<<<nBlocks, 32*EW, 0, stream>>>(msgB, xbB, WrelABB, brel_ab,
                                             WrootABB, broot_ab, out_b, nTiles);
}
